// TextualNetBeta_89919435309116
// MI455X (gfx1250) — matrix operand order repaired and checked by reading, not compiled, not run on hardware
//
#include <hip/hip_runtime.h>
#include <hip/hip_bf16.h>

typedef __attribute__((ext_vector_type(16))) _Float16 v16h;
typedef __attribute__((ext_vector_type(8)))  _Float16 h8v;
typedef __attribute__((ext_vector_type(8)))  float    v8f;

#define BB 128
#define LQ 64
#define OO 7
#define LO 16
#define SS 16
#define WW 32
#define EE 300
#define HH 512
#define HC 256
#define DD 512
#define NHEADS 8
#define DHD 64
#define FFD 2048
#define NEGV   (-1e9f)
#define NEGBIG (-9e15f)

// ---------------------------------------------------------------------------
// Tiled WMMA GEMM: C[M,N] = act( A[M,K] @ B + bias + Res )
// A row-major (lda). B row-major [K,N] (ldb) or, if transB, stored [N,K].
// Block = 256 threads = 8 waves; block tile 128x64; K-step 32.
// LDS tiles are stored in WMMA *fragment order* [subtile][lane][16 halves]
// so each lane's fragment is one 32B contiguous read (2x ds_load_b128).
// Double-buffered LDS: one barrier per K-step, stage-ahead overlaps WMMA.
// All B fragments are preloaded before the WMMA chain so the 8 ds_load_b128
// issue together and waits are staggered instead of dscnt==0 per WMMA.
// ---------------------------------------------------------------------------
__global__ void __launch_bounds__(256)
k_gemm(const float* __restrict__ A, const float* __restrict__ Bm,
       const float* __restrict__ bias, const float* __restrict__ Res,
       float* __restrict__ C,
       int M, int N, int K, int lda, int ldb, int ldres, int ldc,
       int transB, int relu)
{
    // [2 buffers][8 m-subtiles][32 lanes][16 halves]
    __shared__ __align__(32) _Float16 As_f[2][8 * 32 * 16];
    // [2 buffers][4 n-subtiles][32 lanes][16 halves]
    __shared__ __align__(32) _Float16 Bs_f[2][4 * 32 * 16];

    const int tid  = threadIdx.x;
    const int lane = tid & 31;
    const int m0   = blockIdx.y * 128;
    const int n0   = blockIdx.x * 64;
    const bool interior_mn = (m0 + 128 <= M) && (n0 + 64 <= N);

    v8f acc[4] = {{}, {}, {}, {}};

    auto stage = [&](int kk, _Float16* Asp, _Float16* Bsp) {
        const bool fast = interior_mn && (kk + 32 <= K);
        if (fast) {
            // A tile 128x32: 512 slots of 8 halves; 2 slots/thread
#pragma unroll
            for (int ss = 0; ss < 2; ++ss) {
                int s = tid + (ss << 8);
                int m = s >> 2, c0 = (s & 3) << 3;           // c0 in {0,8,16,24}
                const float* src = A + (size_t)(m0 + m) * lda + kk + c0;
                float2 f0 = *(const float2*)(src);
                float2 f1 = *(const float2*)(src + 2);
                float2 f2 = *(const float2*)(src + 4);
                float2 f3 = *(const float2*)(src + 6);
                h8v hv;
                hv[0] = (_Float16)f0.x; hv[1] = (_Float16)f0.y;
                hv[2] = (_Float16)f1.x; hv[3] = (_Float16)f1.y;
                hv[4] = (_Float16)f2.x; hv[5] = (_Float16)f2.y;
                hv[6] = (_Float16)f3.x; hv[7] = (_Float16)f3.y;
                int g  = (c0 >> 3) & 1;                      // lane group
                int h0 = (c0 & 7) + ((c0 >> 4) << 3);        // 0 or 8
                int ls = ((m >> 4) << 5) + (m & 15) + (g << 4);
                *(h8v*)(Asp + (ls << 4) + h0) = hv;
            }
            // B tile 32x64: 256 slots (8 k-halves for one n); 1 slot/thread
            {
                int n = tid & 63, kb = (tid >> 6) << 3;      // kb in {0,8,16,24}
                h8v hv;
                if (transB) {
                    const float* src = Bm + (size_t)(n0 + n) * ldb + kk + kb;
                    float2 f0 = *(const float2*)(src);
                    float2 f1 = *(const float2*)(src + 2);
                    float2 f2 = *(const float2*)(src + 4);
                    float2 f3 = *(const float2*)(src + 6);
                    hv[0] = (_Float16)f0.x; hv[1] = (_Float16)f0.y;
                    hv[2] = (_Float16)f1.x; hv[3] = (_Float16)f1.y;
                    hv[4] = (_Float16)f2.x; hv[5] = (_Float16)f2.y;
                    hv[6] = (_Float16)f3.x; hv[7] = (_Float16)f3.y;
                } else {
#pragma unroll
                    for (int j = 0; j < 8; ++j)
                        hv[j] = (_Float16)Bm[(size_t)(kk + kb + j) * ldb + (n0 + n)];
                }
                int g  = (kb >> 3) & 1, h0 = (kb >> 4) << 3; // h0: 0 or 8
                int ls = ((n >> 4) << 5) + (n & 15) + (g << 4);
                *(h8v*)(Bsp + (ls << 4) + h0) = hv;
            }
        } else {
            // guarded scalar path (edge tiles only)
            for (int ss = 0; ss < 2; ++ss) {
                int s = tid + (ss << 8);
                int m = s >> 2, c0 = (s & 3) << 3;
                for (int j = 0; j < 8; ++j) {
                    int c = c0 + j, gm = m0 + m, gk = kk + c;
                    float v = 0.0f;
                    if (gm < M && gk < K) v = A[(size_t)gm * lda + gk];
                    int g  = (c >> 3) & 1;
                    int h  = (c & 7) + ((c >> 4) << 3);
                    int ls = ((m >> 4) << 5) + (m & 15) + (g << 4);
                    Asp[(ls << 4) + h] = (_Float16)v;
                }
            }
            {
                int n = tid & 63, kb = (tid >> 6) << 3;
                for (int j = 0; j < 8; ++j) {
                    int kc = kb + j, gk = kk + kc, gn = n0 + n;
                    float v = 0.0f;
                    if (gk < K && gn < N)
                        v = transB ? Bm[(size_t)gn * ldb + gk]
                                   : Bm[(size_t)gk * ldb + gn];
                    int g  = (kc >> 3) & 1, h = (kc & 7) + ((kc >> 4) << 3);
                    int ls = ((n >> 4) << 5) + (n & 15) + (g << 4);
                    Bsp[(ls << 4) + h] = (_Float16)v;
                }
            }
        }
    };

    const int nK = (K + 31) >> 5;
    stage(0, As_f[0], Bs_f[0]);
    for (int ki = 0; ki < nK; ++ki) {
        __syncthreads();
        const int cur = ki & 1;
        if (ki + 1 < nK) stage((ki + 1) << 5, As_f[cur ^ 1], Bs_f[cur ^ 1]);
        // each wave: its own 16-row subtile (wv = tid>>5), all four 16-col subtiles.
        // Preload A + all four B fragments, then run the WMMA chain, so the
        // ds_load_b128s issue together and waits stagger (no dscnt==0 stalls).
        const v16h a = *(const v16h*)(As_f[cur] + (tid << 4));
        v16h bfr[4];
#pragma unroll
        for (int t = 0; t < 4; ++t)
            bfr[t] = *(const v16h*)(Bs_f[cur] + (((t << 5) + lane) << 4));
#pragma unroll
        for (int t = 0; t < 4; ++t)
            acc[t] = __builtin_amdgcn_wmma_f32_16x16x32_f16(false, a, false, bfr[t],
                                                            (short)0, acc[t],
                                                            false, false);
    }

    // epilogue: C/D layout -> VGPR r holds M=r (lanes 0-15) / M=r+8 (lanes 16-31)
    const int wv = tid >> 5;
#pragma unroll
    for (int t = 0; t < 4; ++t) {
        int gn = n0 + (t << 4) + (lane & 15);
        if (gn < N) {
#pragma unroll
            for (int r = 0; r < 8; ++r) {
                int gm = m0 + (wv << 4) + r + ((lane >> 4) << 3);
                if (gm < M) {
                    float v = acc[t][r];
                    if (bias) v += bias[gn];
                    if (Res)  v += Res[(size_t)gm * ldres + gn];
                    if (relu) v = fmaxf(v, 0.0f);
                    C[(size_t)gm * ldc + gn] = v;
                }
            }
        }
    }
}

// ---------------------------------------------------------------------------
// Elementwise / reduction helper kernels
// ---------------------------------------------------------------------------
__global__ void k_make_mask(const float* __restrict__ x, int* __restrict__ mask,
                            int rows, int rowlen)
{
    int r = blockIdx.x * blockDim.x + threadIdx.x;
    if (r >= rows) return;
    const float* p = x + (size_t)r * rowlen;
    float s = 0.0f;
    for (int i = 0; i < rowlen; ++i) s += fabsf(p[i]);
    mask[r] = (s == 0.0f) ? 1 : 0;
}

__global__ void k_mask_zero(float* __restrict__ x, const int* __restrict__ mask,
                            int rows, int cols)
{
    int i = blockIdx.x * blockDim.x + threadIdx.x;
    if (i >= rows * cols) return;
    if (mask[i / cols]) x[i] = 0.0f;
}

__global__ void k_lstm_cell(const float* __restrict__ gates, float* __restrict__ c,
                            float* __restrict__ h, float* __restrict__ hs,
                            int Nrow, int Hd, int hs_stride)
{
    int i = blockIdx.x * blockDim.x + threadIdx.x;
    if (i >= Nrow * Hd) return;
    int n = i / Hd, j = i - n * Hd;
    const float* gr = gates + (size_t)n * 4 * Hd;
    float gi = gr[j], gf = gr[Hd + j], gg = gr[2 * Hd + j], go = gr[3 * Hd + j];
    float si = 1.0f / (1.0f + __expf(-gi));
    float sf = 1.0f / (1.0f + __expf(-gf));
    float so = 1.0f / (1.0f + __expf(-go));
    float cn = sf * c[i] + si * tanhf(gg);
    c[i] = cn;
    float hn = so * tanhf(cn);
    h[i] = hn;
    hs[(size_t)n * hs_stride + j] = hn;
}

__global__ void __launch_bounds__(256)
k_layernorm(const float* __restrict__ in, const float* __restrict__ g,
            const float* __restrict__ b, float* __restrict__ out, int D)
{
    int row = blockIdx.x;
    int t = threadIdx.x;
    __shared__ float r1[256], r2[256];
    float s = 0.0f, s2 = 0.0f;
    for (int d = t; d < D; d += 256) {
        float x = in[(size_t)row * D + d];
        s += x; s2 += x * x;
    }
    r1[t] = s; r2[t] = s2;
    __syncthreads();
    for (int st = 128; st > 0; st >>= 1) {
        if (t < st) { r1[t] += r1[t + st]; r2[t] += r2[t + st]; }
        __syncthreads();
    }
    float mean = r1[0] / D;
    float var  = r2[0] / D - mean * mean;
    float inv  = rsqrtf(var + 1e-6f);
    for (int d = t; d < D; d += 256)
        out[(size_t)row * D + d] = (in[(size_t)row * D + d] - mean) * inv * g[d] + b[d];
}

// per-(n,h,l) attention row: scores+softmax+AV, L <= 64
__global__ void __launch_bounds__(64)
k_attn(const float* __restrict__ q, const float* __restrict__ k,
       const float* __restrict__ v, const int* __restrict__ mask,
       float* __restrict__ o, int L)
{
    int l = blockIdx.x, h = blockIdx.y, n = blockIdx.z;
    int t = threadIdx.x;
    __shared__ float qs[DHD];
    __shared__ float sc[64];
    __shared__ float red[64];
    int row = n * L + l;
    qs[t] = q[(size_t)row * DD + h * DHD + t];
    __syncthreads();
    float s = -3.0e38f;
    if (t < L) {
        const float* kp = k + ((size_t)(n * L + t) * DD + h * DHD);
        float acc = 0.0f;
        for (int d = 0; d < DHD; ++d) acc += qs[d] * kp[d];
        s = acc * 0.125f;                 // 1/sqrt(64)
        if (mask[n * L + t]) s = NEGV;
    }
    sc[t] = s; red[t] = s;
    __syncthreads();
    for (int st = 32; st > 0; st >>= 1) {
        if (t < st) red[t] = fmaxf(red[t], red[t + st]);
        __syncthreads();
    }
    float mx = red[0];
    __syncthreads();
    float e = (t < L) ? __expf(sc[t] - mx) : 0.0f;
    sc[t] = e; red[t] = e;
    __syncthreads();
    for (int st = 32; st > 0; st >>= 1) {
        if (t < st) red[t] += red[t + st];
        __syncthreads();
    }
    float inv = 1.0f / red[0];
    __syncthreads();
    float acc = 0.0f;
    for (int m = 0; m < L; ++m)
        acc += sc[m] * v[(size_t)(n * L + m) * DD + h * DHD + t];
    o[(size_t)row * DD + h * DHD + t] = acc * inv;
}

__global__ void k_seq_sum(const float* __restrict__ in, float* __restrict__ out,
                          int G, int L, int D)
{
    int i = blockIdx.x * blockDim.x + threadIdx.x;
    if (i >= G * D) return;
    int g = i / D, d = i - g * D;
    float s = 0.0f;
    for (int l = 0; l < L; ++l) s += in[((size_t)g * L + l) * D + d];
    out[i] = s;
}

__global__ void k_csf_len(const int* __restrict__ cmask, float* __restrict__ len,
                          int n, int W)
{
    int r = blockIdx.x * blockDim.x + threadIdx.x;
    if (r >= n) return;
    int c = 0;
    for (int w = 0; w < W; ++w) c += cmask[r * W + w] ? 0 : 1;
    len[r] = (c == 0) ? NEGBIG : (float)c;
}

__global__ void k_argmax_q2c(const float* __restrict__ qsum, const float* __restrict__ cfw,
                             const float* __restrict__ len, int* __restrict__ ix)
{
    int b = blockIdx.x * blockDim.x + threadIdx.x;
    if (b >= BB) return;
    float best = -3.0e38f; int bi = 0;
    for (int s = 0; s < SS; ++s) {
        float acc = 0.0f;
        const float* cp = cfw + (size_t)(b * SS + s) * DD;
        for (int d = 0; d < DD; ++d) acc += qsum[(size_t)b * DD + d] * cp[d];
        float sc = acc / len[b * SS + s];
        if (sc > best) { best = sc; bi = s; }
    }
    ix[b] = bi;
}

__global__ void k_argmax_o2c(const float* __restrict__ osum, const float* __restrict__ cfw,
                             const float* __restrict__ len, int* __restrict__ ix)
{
    int i = blockIdx.x * blockDim.x + threadIdx.x;
    if (i >= BB * OO) return;
    int b = i / OO;
    float best = -3.0e38f; int bi = 0;
    for (int s = 0; s < SS; ++s) {
        float acc = 0.0f;
        const float* cp = cfw + (size_t)(b * SS + s) * DD;
        for (int d = 0; d < DD; ++d) acc += osum[(size_t)i * DD + d] * cp[d];
        float sc = acc / len[b * SS + s];
        if (sc > best) { best = sc; bi = s; }
    }
    ix[i] = bi;
}

__global__ void k_build_ctx(const float* __restrict__ cf, const int* __restrict__ ixq,
                            const int* __restrict__ ixo, const int* __restrict__ osum,
                            float* __restrict__ ctx)
{
    int i = blockIdx.x * blockDim.x + threadIdx.x;
    int total = BB * OO * 64 * DD;
    if (i >= total) return;
    int d = i % DD;
    int w = (i / DD) % 64;
    int o = (i / (DD * 64)) % OO;
    int b = i / (DD * 64 * OO);
    float v = 0.0f;
    if (!osum[b * OO + o]) {
        if (w < WW)
            v = cf[(((size_t)b * SS + ixq[b]) * WW + w) * DD + d];        // que_csf
        else
            v = cf[(((size_t)ixo[b * OO + o]) * WW + (w - WW)) * DD + d]; // cf[0][ix_o2c]
    }
    ctx[i] = v;
}

__global__ void k_row_dot(const float* __restrict__ x, const float* __restrict__ w,
                          const float* __restrict__ bias, float* __restrict__ out,
                          int rows, int K)
{
    int r = blockIdx.x * blockDim.x + threadIdx.x;
    if (r >= rows) return;
    float s = bias ? bias[0] : 0.0f;
    const float* p = x + (size_t)r * K;
    for (int i = 0; i < K; ++i) s += p[i] * w[i];
    out[r] = s;
}

__global__ void __launch_bounds__(256)
k_attflat_pool(const float* __restrict__ x, const float* __restrict__ a,
               const int* __restrict__ mask, float* __restrict__ pooled,
               int L, int D)
{
    int g = blockIdx.x;
    int t = threadIdx.x;
    __shared__ float sa_[64];
    __shared__ float red[64];
    if (t < 64) {
        float s = -3.0e38f;
        if (t < L) {
            s = a[g * L + t];
            if (mask[g * L + t]) s = NEGV;
        }
        sa_[t] = s; red[t] = s;
    }
    __syncthreads();
    for (int st = 32; st > 0; st >>= 1) {
        if (t < st) red[t] = fmaxf(red[t], red[t + st]);
        __syncthreads();
    }
    float mx = red[0];
    __syncthreads();
    if (t < 64) {
        float e = (t < L) ? __expf(sa_[t] - mx) : 0.0f;
        sa_[t] = e; red[t] = e;
    }
    __syncthreads();
    for (int st = 32; st > 0; st >>= 1) {
        if (t < st) red[t] += red[t + st];
        __syncthreads();
    }
    float inv = 1.0f / red[0];
    __syncthreads();
    for (int d = t; d < D; d += 256) {
        float s = 0.0f;
        for (int l = 0; l < L; ++l) s += sa_[l] * x[((size_t)g * L + l) * D + d];
        pooled[(size_t)g * D + d] = s * inv;
    }
}

__global__ void k_final(const float* __restrict__ fq, const float* __restrict__ fo,
                        const float* __restrict__ fc, const float* __restrict__ clsW,
                        const float* __restrict__ clsb, const int* __restrict__ osum,
                        float* __restrict__ out)
{
    int i = blockIdx.x * blockDim.x + threadIdx.x;
    if (i >= BB * OO) return;
    int b = i / OO;
    float s = clsb[0];
    for (int d = 0; d < DD; ++d) s += clsW[d]          * fq[(size_t)b * DD + d];
    for (int d = 0; d < DD; ++d) s += clsW[DD + d]     * fo[(size_t)i * DD + d];
    for (int d = 0; d < DD; ++d) s += clsW[2 * DD + d] * fc[(size_t)i * DD + d];
    out[i] = osum[i] ? NEGBIG : s;
}

// ---------------------------------------------------------------------------
// Host-side orchestration helpers
// ---------------------------------------------------------------------------
static void gemm(hipStream_t st, const float* A, const float* B, const float* bias,
                 const float* res, float* C, int M, int N, int K,
                 int lda, int ldb, int ldres, int ldc, int transB, int relu)
{
    dim3 grid((N + 63) / 64, (M + 127) / 128);
    k_gemm<<<grid, 256, 0, st>>>(A, B, bias, res, C, M, N, K,
                                 lda, ldb, ldres, ldc, transB, relu);
}

static void run_lstm(hipStream_t st, const float* X, int Nseq, int T, int Ein,
                     const float* Wih, const float* Whh, const float* bvec, int Hd,
                     float* hs_base, int hs_rowstride, int hs_tstride, int reverse,
                     float* xproj, float* gtmp, float* hb, float* cb)
{
    int G4 = 4 * Hd;
    // all-timestep input projection: [Nseq*T, E] @ Wih^T + b
    gemm(st, X, Wih, bvec, nullptr, xproj, Nseq * T, G4, Ein, Ein, Ein, 0, G4, 1, 0);
    hipMemsetAsync(hb, 0, (size_t)Nseq * Hd * sizeof(float), st);
    hipMemsetAsync(cb, 0, (size_t)Nseq * Hd * sizeof(float), st);
    for (int s = 0; s < T; ++s) {
        int t = reverse ? (T - 1 - s) : s;
        gemm(st, hb, Whh, nullptr, xproj + (size_t)t * G4, gtmp,
             Nseq, G4, Hd, Hd, Hd, T * G4, G4, 1, 0);
        int tot = Nseq * Hd;
        k_lstm_cell<<<(tot + 255) / 256, 256, 0, st>>>(
            gtmp, cb, hb, hs_base + (size_t)t * hs_tstride, Nseq, Hd, hs_rowstride);
    }
}

static void sa_run(hipStream_t st, float* x, const int* mask, int G, int L,
                   void* const* din, int pb,
                   float* q, float* k, float* v, float* o, float* f1)
{
    const float* Wq  = (const float*)din[pb + 0];  const float* bq  = (const float*)din[pb + 1];
    const float* Wk  = (const float*)din[pb + 2];  const float* bk  = (const float*)din[pb + 3];
    const float* Wv  = (const float*)din[pb + 4];  const float* bv  = (const float*)din[pb + 5];
    const float* Wo  = (const float*)din[pb + 6];  const float* bo  = (const float*)din[pb + 7];
    const float* l1g = (const float*)din[pb + 8];  const float* l1b = (const float*)din[pb + 9];
    const float* Wf1 = (const float*)din[pb + 10]; const float* bf1 = (const float*)din[pb + 11];
    const float* Wf2 = (const float*)din[pb + 12]; const float* bf2 = (const float*)din[pb + 13];
    const float* l2g = (const float*)din[pb + 14]; const float* l2b = (const float*)din[pb + 15];

    int Nt = G * L;
    gemm(st, x, Wq, bq, nullptr, q, Nt, DD, DD, DD, DD, 0, DD, 0, 0);
    gemm(st, x, Wk, bk, nullptr, k, Nt, DD, DD, DD, DD, 0, DD, 0, 0);
    gemm(st, x, Wv, bv, nullptr, v, Nt, DD, DD, DD, DD, 0, DD, 0, 0);
    dim3 ag(L, NHEADS, G);
    k_attn<<<ag, 64, 0, st>>>(q, k, v, mask, o, L);
    // t1 = o@Wo + bo + x   (reuse q)
    gemm(st, o, Wo, bo, x, q, Nt, DD, DD, DD, DD, DD, DD, 0, 0);
    // x1 = LN(t1)          (reuse k)
    k_layernorm<<<Nt, 256, 0, st>>>(q, l1g, l1b, k, DD);
    // FFN, chunked rows    (f2 -> reuse v)
    const int CH = 16384;
    for (int r0 = 0; r0 < Nt; r0 += CH) {
        int mr = (Nt - r0 < CH) ? (Nt - r0) : CH;
        gemm(st, k + (size_t)r0 * DD, Wf1, bf1, nullptr, f1, mr, FFD, DD, DD, FFD, 0, FFD, 0, 1);
        gemm(st, f1, Wf2, bf2, k + (size_t)r0 * DD, v + (size_t)r0 * DD,
             mr, DD, FFD, FFD, DD, DD, DD, 0, 0);
    }
    k_layernorm<<<Nt, 256, 0, st>>>(v, l2g, l2b, x, DD);
}

static void attflat_run(hipStream_t st, const float* x, const int* mask, int G, int L,
                        void* const* din, float* out,
                        float* tbuf, float* abuf, float* pbuf)
{
    const float* W1 = (const float*)din[44]; const float* b1 = (const float*)din[45];
    const float* W2 = (const float*)din[46]; const float* b2 = (const float*)din[47];
    const float* Wm = (const float*)din[48]; const float* bm = (const float*)din[49];
    int Nt = G * L;
    gemm(st, x, W1, b1, nullptr, tbuf, Nt, DD, DD, DD, DD, 0, DD, 0, 1);
    k_row_dot<<<(Nt + 255) / 256, 256, 0, st>>>(tbuf, W2, b2, abuf, Nt, DD);
    k_attflat_pool<<<G, 256, 0, st>>>(x, abuf, mask, pbuf, L, DD);
    gemm(st, pbuf, Wm, bm, nullptr, out, G, DD, DD, DD, DD, 0, DD, 0, 0);
}

// ---------------------------------------------------------------------------
extern "C" void kernel_launch(void* const* d_in, const int* in_sizes, int n_in,
                              void* d_out, int out_size, void* d_ws, size_t ws_size,
                              hipStream_t stream)
{
    (void)in_sizes; (void)n_in; (void)out_size; (void)ws_size;

    const float* que     = (const float*)d_in[0];
    const float* opt     = (const float*)d_in[1];
    const float* csf     = (const float*)d_in[2];
    const float* qo_Wih  = (const float*)d_in[3];
    const float* qo_Whh  = (const float*)d_in[4];
    const float* qo_b    = (const float*)d_in[5];
    const float* csWih_f = (const float*)d_in[6];
    const float* csWhh_f = (const float*)d_in[7];
    const float* csb_f   = (const float*)d_in[8];
    const float* csWih_b = (const float*)d_in[9];
    const float* csWhh_b = (const float*)d_in[10];
    const float* csb_b   = (const float*)d_in[11];
    // qo_att: d_in[12..27], cs_att: d_in[28..43], flat: d_in[44..49]
    const float* clsW    = (const float*)d_in[50];
    const float* clsb    = (const float*)d_in[51];

    char* base = (char*)d_ws;
    size_t off = 0;
    auto af = [&](size_t n) -> float* {
        float* p = (float*)(base + off);
        off += ((n * sizeof(float) + 255ull) & ~255ull);
        return p;
    };
    auto ai = [&](size_t n) -> int* {
        int* p = (int*)(base + off);
        off += ((n * sizeof(int) + 255ull) & ~255ull);
        return p;
    };

    // persistent buffers
    float* qf      = af((size_t)BB * LQ * DD);
    float* of      = af((size_t)BB * OO * LO * DD);
    float* cf      = af((size_t)BB * SS * WW * DD);
    float* ctx     = af((size_t)BB * OO * 64 * DD);
    float* flat_qb = af((size_t)BB * DD);
    float* flat_o  = af((size_t)BB * OO * DD);
    float* flat_c  = af((size_t)BB * OO * DD);
    float* cfw     = af((size_t)BB * SS * DD);
    float* qsum    = af((size_t)BB * DD);
    float* osum_f  = af((size_t)BB * OO * DD);
    float* clen    = af((size_t)BB * SS);
    int* qmask   = ai((size_t)BB * LQ);
    int* omask   = ai((size_t)BB * OO * LO);
    int* cmask   = ai((size_t)BB * SS * WW);
    int* osum    = ai((size_t)BB * OO);
    int* ctxmask = ai((size_t)BB * OO * 64);
    int* ixq     = ai(BB);
    int* ixo     = ai((size_t)BB * OO);
    const size_t scratch0 = off;

    // masks
    k_make_mask<<<(BB * LQ + 255) / 256, 256, 0, stream>>>(que, qmask, BB * LQ, EE);
    k_make_mask<<<(BB * OO * LO + 255) / 256, 256, 0, stream>>>(opt, omask, BB * OO * LO, EE);
    k_make_mask<<<(BB * SS * WW + 255) / 256, 256, 0, stream>>>(csf, cmask, BB * SS * WW, EE);
    k_make_mask<<<(BB * OO + 255) / 256, 256, 0, stream>>>(opt, osum, BB * OO, LO * EE);

    // ---- question encoder ----
    off = scratch0;
    {
        float* xproj = af((size_t)BB * LQ * 4 * HH);
        float* gtmp  = af((size_t)BB * 4 * HH);
        float* hb    = af((size_t)BB * HH);
        float* cb    = af((size_t)BB * HH);
        run_lstm(stream, que, BB, LQ, EE, qo_Wih, qo_Whh, qo_b, HH,
                 qf, LQ * DD, DD, 0, xproj, gtmp, hb, cb);
    }
    k_mask_zero<<<(BB * LQ * DD + 255) / 256, 256, 0, stream>>>(qf, qmask, BB * LQ, DD);
    off = scratch0;
    {
        int Nt = BB * LQ;
        float* q  = af((size_t)Nt * DD);
        float* k_ = af((size_t)Nt * DD);
        float* v  = af((size_t)Nt * DD);
        float* o  = af((size_t)Nt * DD);
        float* f1 = af((size_t)((Nt < 16384) ? Nt : 16384) * FFD);
        sa_run(stream, qf, qmask, BB, LQ, d_in, 12, q, k_, v, o, f1);
    }
    k_mask_zero<<<(BB * LQ * DD + 255) / 256, 256, 0, stream>>>(qf, qmask, BB * LQ, DD);

    // ---- option encoder (shared qo weights) ----
    off = scratch0;
    {
        int Nseq = BB * OO;
        float* xproj = af((size_t)Nseq * LO * 4 * HH);
        float* gtmp  = af((size_t)Nseq * 4 * HH);
        float* hb    = af((size_t)Nseq * HH);
        float* cb    = af((size_t)Nseq * HH);
        run_lstm(stream, opt, Nseq, LO, EE, qo_Wih, qo_Whh, qo_b, HH,
                 of, LO * DD, DD, 0, xproj, gtmp, hb, cb);
    }
    k_mask_zero<<<(BB * OO * LO * DD + 255) / 256, 256, 0, stream>>>(of, omask, BB * OO * LO, DD);
    off = scratch0;
    {
        int Nt = BB * OO * LO;
        float* q  = af((size_t)Nt * DD);
        float* k_ = af((size_t)Nt * DD);
        float* v  = af((size_t)Nt * DD);
        float* o  = af((size_t)Nt * DD);
        float* f1 = af((size_t)((Nt < 16384) ? Nt : 16384) * FFD);
        sa_run(stream, of, omask, BB * OO, LO, d_in, 12, q, k_, v, o, f1);
    }
    k_mask_zero<<<(BB * OO * LO * DD + 255) / 256, 256, 0, stream>>>(of, omask, BB * OO * LO, DD);

    // ---- closest-sentence bi-LSTM encoder ----
    off = scratch0;
    {
        int Nseq = BB * SS;
        float* xproj = af((size_t)Nseq * WW * 4 * HC);
        float* gtmp  = af((size_t)Nseq * 4 * HC);
        float* hb    = af((size_t)Nseq * HC);
        float* cb    = af((size_t)Nseq * HC);
        run_lstm(stream, csf, Nseq, WW, EE, csWih_f, csWhh_f, csb_f, HC,
                 cf, WW * DD, DD, 0, xproj, gtmp, hb, cb);
        run_lstm(stream, csf, Nseq, WW, EE, csWih_b, csWhh_b, csb_b, HC,
                 cf + HC, WW * DD, DD, 1, xproj, gtmp, hb, cb);
    }
    k_mask_zero<<<(BB * SS * WW * DD + 255) / 256, 256, 0, stream>>>(cf, cmask, BB * SS * WW, DD);
    off = scratch0;
    {
        int Nt = BB * SS * WW;
        float* q  = af((size_t)Nt * DD);
        float* k_ = af((size_t)Nt * DD);
        float* v  = af((size_t)Nt * DD);
        float* o  = af((size_t)Nt * DD);
        float* f1 = af((size_t)16384 * FFD);
        sa_run(stream, cf, cmask, BB * SS, WW, d_in, 28, q, k_, v, o, f1);
    }
    k_mask_zero<<<(BB * SS * WW * DD + 255) / 256, 256, 0, stream>>>(cf, cmask, BB * SS * WW, DD);

    // ---- find_context ----
    k_seq_sum<<<(BB * SS * DD + 255) / 256, 256, 0, stream>>>(cf, cfw, BB * SS, WW, DD);
    k_seq_sum<<<(BB * DD + 255) / 256, 256, 0, stream>>>(qf, qsum, BB, LQ, DD);
    k_seq_sum<<<(BB * OO * DD + 255) / 256, 256, 0, stream>>>(of, osum_f, BB * OO, LO, DD);
    k_csf_len<<<(BB * SS + 255) / 256, 256, 0, stream>>>(cmask, clen, BB * SS, WW);
    k_argmax_q2c<<<(BB + 127) / 128, 128, 0, stream>>>(qsum, cfw, clen, ixq);
    k_argmax_o2c<<<(BB * OO + 127) / 128, 128, 0, stream>>>(osum_f, cfw, clen, ixo);
    k_build_ctx<<<(BB * OO * 64 * DD + 255) / 256, 256, 0, stream>>>(cf, ixq, ixo, osum, ctx);
    k_make_mask<<<(BB * OO * 64 + 255) / 256, 256, 0, stream>>>(ctx, ctxmask, BB * OO * 64, DD);

    // ---- att_flat x3 ----
    off = scratch0;
    {
        float* tbuf = af((size_t)BB * OO * 64 * DD);
        float* abuf = af((size_t)BB * OO * 64);
        float* pbuf = af((size_t)BB * OO * DD);
        attflat_run(stream, qf, qmask, BB, LQ, d_in, flat_qb, tbuf, abuf, pbuf);
        attflat_run(stream, of, omask, BB * OO, LO, d_in, flat_o, tbuf, abuf, pbuf);
        attflat_run(stream, ctx, ctxmask, BB * OO, 64, d_in, flat_c, tbuf, abuf, pbuf);
    }

    // ---- classifier ----
    k_final<<<(BB * OO + 255) / 256, 256, 0, stream>>>(
        flat_qb, flat_o, flat_c, clsW, clsb, osum, (float*)d_out);
}
